// Inpainting_76098230550871
// MI455X (gfx1250) — matrix operand order repaired and checked by reading, not compiled, not run on hardware
//
#include <hip/hip_runtime.h>
#include <hip/hip_bf16.h>
#include <math.h>

typedef __attribute__((ext_vector_type(16))) __bf16 v16bf;
typedef __attribute__((ext_vector_type(8)))  float  v8f;

#define HW 16384L
#define IMG_H 128
#define IMG_W 128
#define NB 4

// db3 analysis filters
__device__ __constant__ float c_lo[6] = {0.035226291882100656f, -0.08544127388224149f,
                                         -0.13501102001039084f, 0.4598775021193313f,
                                         0.8068915093133388f,   0.3326705529509569f};
__device__ __constant__ float c_hi[6] = {-0.3326705529509569f, 0.8068915093133388f,
                                         -0.4598775021193313f, -0.13501102001039084f,
                                         0.08544127388224149f, 0.035226291882100656f};

// ---------------------------------------------------------------------------
// Channel LayerNorm over C=192 per (b, pixel)
// ---------------------------------------------------------------------------
__global__ __launch_bounds__(256) void ln_kernel(const float* __restrict__ x,
                                                 const float* __restrict__ g,
                                                 const float* __restrict__ be,
                                                 float* __restrict__ out) {
    long p = (long)blockIdx.x * blockDim.x + threadIdx.x;   // 0 .. B*HW-1
    long b = p >> 14;
    long sp = p & (HW - 1);
    const float* xb = x + b * 192 * HW + sp;
    float s = 0.f, s2 = 0.f;
    #pragma unroll 4
    for (int c = 0; c < 192; ++c) {
        float v = xb[(long)c * HW];
        s += v; s2 += v * v;
    }
    float mu = s * (1.f / 192.f);
    float var = s2 * (1.f / 192.f) - mu * mu;
    float rstd = rsqrtf(var + 1e-5f);
    float* ob = out + b * 192 * HW + sp;
    #pragma unroll 4
    for (int c = 0; c < 192; ++c)
        ob[(long)c * HW] = (xb[(long)c * HW] - mu) * rstd * g[c] + be[c];
}

// ---------------------------------------------------------------------------
// Generic depthwise 3x3, pad 1 (NCHW)
// ---------------------------------------------------------------------------
__global__ __launch_bounds__(256) void dw3x3_kernel(const float* __restrict__ in,
                                                    const float* __restrict__ w,
                                                    float* __restrict__ out,
                                                    int Ch, int H, int W, long total) {
    long idx = (long)blockIdx.x * blockDim.x + threadIdx.x;
    if (idx >= total) return;
    int xo = (int)(idx % W);
    long t = idx / W;
    int yo = (int)(t % H);
    long bc = t / H;                 // b*Ch + c
    int c = (int)(bc % Ch);
    const float* wp = w + (long)c * 9;
    const float* ip = in + bc * (long)H * W;
    float s = 0.f;
    #pragma unroll
    for (int ky = 0; ky < 3; ++ky) {
        int yy = yo + ky - 1;
        if (yy < 0 || yy >= H) continue;
        #pragma unroll
        for (int kx = 0; kx < 3; ++kx) {
            int xx = xo + kx - 1;
            if (xx < 0 || xx >= W) continue;
            s += wp[ky * 3 + kx] * ip[(long)yy * W + xx];
        }
    }
    out[idx] = s;
}

// ---------------------------------------------------------------------------
// db3 DWT: pad 4, stride 2, reversed filters -> 4 bands (B,C,66,66)
// ---------------------------------------------------------------------------
__global__ __launch_bounds__(256) void dwt_kernel(const float* __restrict__ xn,
                                                  float* __restrict__ bands, long total) {
    long idx = (long)blockIdx.x * blockDim.x + threadIdx.x;
    if (idx >= total) return;
    int ox = (int)(idx % 66);
    long t = idx / 66;
    int oy = (int)(t % 66);
    long bc = t / 66;                // b*192 + c
    const float* ip = xn + bc * HW;
    int yb = 2 * oy - 4, xb = 2 * ox - 4;
    float rl[6], rh[6];
    #pragma unroll
    for (int i = 0; i < 6; ++i) {
        int yy = yb + i;
        float al = 0.f, ah = 0.f;
        if (yy >= 0 && yy < IMG_H) {
            #pragma unroll
            for (int j = 0; j < 6; ++j) {
                int xx = xb + j;
                if (xx < 0 || xx >= IMG_W) continue;
                float v = ip[(long)yy * IMG_W + xx];
                al += c_lo[5 - j] * v;       // reversed filters
                ah += c_hi[5 - j] * v;
            }
        }
        rl[i] = al; rh[i] = ah;
    }
    float ll = 0.f, lh = 0.f, hl = 0.f, hh = 0.f;
    #pragma unroll
    for (int i = 0; i < 6; ++i) {
        ll += c_lo[5 - i] * rl[i];
        lh += c_lo[5 - i] * rh[i];
        hl += c_hi[5 - i] * rl[i];
        hh += c_hi[5 - i] * rh[i];
    }
    long BSZ = total;                // 4*192*66*66 per band
    bands[0 * BSZ + idx] = ll;
    bands[1 * BSZ + idx] = lh;
    bands[2 * BSZ + idx] = hl;
    bands[3 * BSZ + idx] = hh;
}

// ---------------------------------------------------------------------------
// db3 IDWT: zero-insert upsample x2, pad 1, 6x6 correlation, sum of 4 bands
// ---------------------------------------------------------------------------
__global__ __launch_bounds__(256) void idwt_kernel(const float* __restrict__ bands,
                                                   float* __restrict__ q, long total, long BSZ) {
    long idx = (long)blockIdx.x * blockDim.x + threadIdx.x;
    if (idx >= total) return;
    int xo = (int)(idx % IMG_W);
    long t = idx / IMG_W;
    int yo = (int)(t % IMG_H);
    long bc = t / IMG_H;             // b*192 + c
    float s = 0.f;
    #pragma unroll
    for (int i = 0; i < 6; ++i) {
        int u = yo - 1 + i;
        if (u & 1) continue;
        int uu = u >> 1;
        if (uu < 0 || uu >= 66) continue;
        #pragma unroll
        for (int j = 0; j < 6; ++j) {
            int v = xo - 1 + j;
            if (v & 1) continue;
            int vv = v >> 1;
            if (vv < 0 || vv >= 66) continue;
            long bix = bc * 4356 + (long)uu * 66 + vv;
            float li = c_lo[i], hi = c_hi[i], lj = c_lo[j], hj = c_hi[j];
            s += li * lj * bands[0 * BSZ + bix];
            s += li * hj * bands[1 * BSZ + bix];
            s += hi * lj * bands[2 * BSZ + bix];
            s += hi * hj * bands[3 * BSZ + bix];
        }
    }
    q[idx] = s;
}

// ---------------------------------------------------------------------------
// Row reciprocal-norm: one workgroup per row of length 16384
// ---------------------------------------------------------------------------
__global__ __launch_bounds__(256) void rownorm_kernel(const float* __restrict__ base,
                                                      long bStride, int rowsPerBatch,
                                                      float* __restrict__ out) {
    int row = blockIdx.x;
    int b = row / rowsPerBatch, r = row % rowsPerBatch;
    const float* p = base + (long)b * bStride + (long)r * HW;
    float s = 0.f;
    for (int i = threadIdx.x; i < (int)HW; i += 256) {
        float v = p[i];
        s += v * v;
    }
    __shared__ float red[8];
    for (int off = 16; off > 0; off >>= 1) s += __shfl_down(s, off, 32);
    if ((threadIdx.x & 31) == 0) red[threadIdx.x >> 5] = s;
    __syncthreads();
    if (threadIdx.x == 0) {
        float tt = 0.f;
        #pragma unroll
        for (int i = 0; i < 8; ++i) tt += red[i];
        out[row] = 1.f / fmaxf(sqrtf(tt), 1e-12f);
    }
}

// ---------------------------------------------------------------------------
// Gram (q k^T over n=16384) with bf16 WMMA, K split over 8 waves,
// LDS float-atomic reduce, fold 1/||q||*1/||k||*temp, row softmax.
// One workgroup per (b,h).
// ---------------------------------------------------------------------------
__global__ __launch_bounds__(256) void gram_softmax_kernel(const float* __restrict__ q0,
                                                           const float* __restrict__ kv0,
                                                           const float* __restrict__ rq,
                                                           const float* __restrict__ rk,
                                                           const float* __restrict__ temp,
                                                           float* __restrict__ attn) {
    int bh = blockIdx.x;
    int b = bh >> 2, h = bh & 3;
    const float* Q = q0 + ((long)b * 192 + h * 48) * HW;
    const float* K = kv0 + ((long)b * 384 + h * 48) * HW;   // k half of kv

    __shared__ float S[48][49];
    int tid = threadIdx.x;
    for (int i = tid; i < 48 * 49; i += 256) ((float*)S)[i] = 0.f;
    __syncthreads();

    int wave = tid >> 5, lane = tid & 31;
    int half = lane >> 4, l16 = lane & 15;
    long kBase = (long)wave * 2048;

    v8f acc[3][3] = {};
    for (int kk = 0; kk < 2048; kk += 32) {
        long nb = kBase + kk;
        v16bf af[3], bfr[3];
        #pragma unroll
        for (int tI = 0; tI < 3; ++tI) {
            const float* qr = Q + (long)(tI * 16 + l16) * HW + nb;
            const float* kr = K + (long)(tI * 16 + l16) * HW + nb;
            #pragma unroll
            for (int e = 0; e < 16; ++e) af[tI][e] = (__bf16)qr[half * 16 + e];
            #pragma unroll
            for (int e = 0; e < 16; ++e) bfr[tI][e] = (__bf16)kr[half * 16 + e];
        }
        #pragma unroll
        for (int i = 0; i < 3; ++i)
            #pragma unroll
            for (int j = 0; j < 3; ++j)
                acc[i][j] = __builtin_amdgcn_wmma_f32_16x16x32_bf16(
                    false, af[i], false, bfr[j], (short)0, acc[i][j], false, false);
    }
    #pragma unroll
    for (int i = 0; i < 3; ++i)
        #pragma unroll
        for (int j = 0; j < 3; ++j)
            #pragma unroll
            for (int r = 0; r < 8; ++r)
                atomicAdd(&S[i * 16 + r + 8 * half][j * 16 + l16], acc[i][j][r]);
    __syncthreads();

    if (tid < 48) {
        int d = tid;
        float sq = rq[b * 192 + h * 48 + d];
        float tp = temp[h];
        float row[48];
        float mx = -3.4e38f;
        #pragma unroll 8
        for (int e = 0; e < 48; ++e) {
            float v = S[d][e] * sq * rk[b * 192 + h * 48 + e] * tp;
            row[e] = v;
            mx = fmaxf(mx, v);
        }
        float sum = 0.f;
        #pragma unroll 8
        for (int e = 0; e < 48; ++e) {
            row[e] = __expf(row[e] - mx);
            sum += row[e];
        }
        float inv = 1.f / sum;
        float* ap = attn + ((long)bh * 48 + d) * 48;
        #pragma unroll 8
        for (int e = 0; e < 48; ++e) ap[e] = row[e] * inv;
    }
}

// ---------------------------------------------------------------------------
// Generic bf16-WMMA GEMM: Out[co,n] = sum_ci W[co,ci] * X[ci,n] (+residual)
// Tiles: BM=64, BN=128, BK=32; 8 waves, each 2x2 of 16x16x32 WMMA tiles.
// Dual batch strides (outer/inner) so grid.z can enumerate (b) or (b,h).
// ---------------------------------------------------------------------------
#define GBM 64
#define GBN 128
#define GBK 32
#define APD 40   // padded LDS A row stride (elements)
#define BPD 40   // padded LDS B row stride (elements)

__global__ __launch_bounds__(256) void gemm_bf16_kernel(
    const float* __restrict__ W0, const float* __restrict__ X0,
    float* __restrict__ O0, const float* __restrict__ R0,
    int Cout, int Cin, int Npix,
    long wSO, long wSI, long xSO, long xSI,
    long oSO, long oSI, long rSO, long rSI, int innerCount) {
    int z = blockIdx.z;
    int zo = z / innerCount, zi = z % innerCount;
    const float* Wp = W0 + (long)zo * wSO + (long)zi * wSI;
    const float* Xp = X0 + (long)zo * xSO + (long)zi * xSI;
    float* Op = O0 + (long)zo * oSO + (long)zi * oSI;
    const float* Rp = R0 ? (R0 + (long)zo * rSO + (long)zi * rSI) : nullptr;

    int coBase = blockIdx.y * GBM;
    int nBase = blockIdx.x * GBN;

    __shared__ __bf16 ldsA[GBM * APD];
    __shared__ __bf16 ldsB[GBN * BPD];

    int tid = threadIdx.x;
    int wave = tid >> 5, lane = tid & 31;
    int half = lane >> 4, l16 = lane & 15;
    int wm = wave >> 2;   // 0..1 -> 32 rows each
    int wn = wave & 3;    // 0..3 -> 32 cols each

    v8f acc[2][2] = {};

    for (int k0 = 0; k0 < Cin; k0 += GBK) {
        // Stage A tile (GBM x GBK) as bf16
        #pragma unroll
        for (int i = 0; i < (GBM * GBK) / 256; ++i) {
            int lin = tid + i * 256;
            int m = lin >> 5, kk = lin & 31;
            int gm = coBase + m, gk = k0 + kk;
            float v = (gm < Cout && gk < Cin) ? Wp[(long)gm * Cin + gk] : 0.f;
            ldsA[m * APD + kk] = (__bf16)v;
        }
        // Stage B tile (GBK x GBN) transposed as bf16: ldsB[n][kk]
        #pragma unroll
        for (int i = 0; i < (GBN * GBK) / 256; ++i) {
            int lin = tid + i * 256;
            int n = lin & 127, kk = lin >> 7;
            int gk = k0 + kk;
            float v = (gk < Cin) ? Xp[(long)gk * Npix + nBase + n] : 0.f;
            ldsB[n * BPD + kk] = (__bf16)v;
        }
        // Prefetch next X tile into L2 (global_prefetch_b8)
        if (k0 + GBK < Cin)
            __builtin_prefetch(&Xp[(long)(k0 + GBK + (tid >> 7)) * Npix + nBase + (tid & 127)], 0, 0);
        __syncthreads();

        v16bf afrag[2], bfrag[2];
        #pragma unroll
        for (int tm = 0; tm < 2; ++tm) {
            int row = wm * 32 + tm * 16 + l16;
            #pragma unroll
            for (int e = 0; e < 16; ++e) afrag[tm][e] = ldsA[row * APD + half * 16 + e];
        }
        #pragma unroll
        for (int tn = 0; tn < 2; ++tn) {
            int col = wn * 32 + tn * 16 + l16;
            #pragma unroll
            for (int e = 0; e < 16; ++e) bfrag[tn][e] = ldsB[col * BPD + half * 16 + e];
        }
        #pragma unroll
        for (int tm = 0; tm < 2; ++tm)
            #pragma unroll
            for (int tn = 0; tn < 2; ++tn)
                acc[tm][tn] = __builtin_amdgcn_wmma_f32_16x16x32_bf16(
                    false, afrag[tm], false, bfrag[tn], (short)0, acc[tm][tn], false, false);
        __syncthreads();
    }

    // Epilogue: C/D layout VGPR r -> M=r (lanes 0-15), M=r+8 (lanes 16-31)
    #pragma unroll
    for (int tm = 0; tm < 2; ++tm)
        #pragma unroll
        for (int tn = 0; tn < 2; ++tn) {
            int n = nBase + wn * 32 + tn * 16 + l16;
            #pragma unroll
            for (int r = 0; r < 8; ++r) {
                int m = coBase + wm * 32 + tm * 16 + r + 8 * half;
                if (m < Cout) {
                    long oix = (long)m * Npix + n;
                    float v = acc[tm][tn][r];
                    if (Rp) v += Rp[oix];
                    Op[oix] = v;
                }
            }
        }
}

// ---------------------------------------------------------------------------
// GDFN gate: depthwise 3x3 on channels j and 510+j, gelu(y1)*y2
// ---------------------------------------------------------------------------
__global__ __launch_bounds__(256) void gate_kernel(const float* __restrict__ hid,
                                                   const float* __restrict__ wdw,
                                                   float* __restrict__ out, long total) {
    long idx = (long)blockIdx.x * blockDim.x + threadIdx.x;
    if (idx >= total) return;
    int xo = (int)(idx % IMG_W);
    long t = idx / IMG_W;
    int yo = (int)(t % IMG_H);
    long bj = t / IMG_H;             // b*510 + j
    int j = (int)(bj % 510);
    long b = bj / 510;
    const float* i1 = hid + (b * 1020 + j) * HW;
    const float* i2 = hid + (b * 1020 + 510 + j) * HW;
    const float* w1 = wdw + (long)j * 9;
    const float* w2 = wdw + (long)(510 + j) * 9;
    float y1 = 0.f, y2 = 0.f;
    #pragma unroll
    for (int ky = 0; ky < 3; ++ky) {
        int yy = yo + ky - 1;
        if (yy < 0 || yy >= IMG_H) continue;
        #pragma unroll
        for (int kx = 0; kx < 3; ++kx) {
            int xx = xo + kx - 1;
            if (xx < 0 || xx >= IMG_W) continue;
            long o = (long)yy * IMG_W + xx;
            y1 += w1[ky * 3 + kx] * i1[o];
            y2 += w2[ky * 3 + kx] * i2[o];
        }
    }
    float g = 0.5f * y1 * (1.f + erff(y1 * 0.70710678118654752f));
    out[(b * 510 + j) * HW + (long)yo * IMG_W + xo] = g * y2;
}

// ---------------------------------------------------------------------------
// Workspace layout (float offsets)
// ---------------------------------------------------------------------------
static const long A_OFF = 0;                       // 1020*HW*B (qkv_out, bands, ffn hidden)
static const long BAND_RAW = 0;                    // 4 bands * 4*192*4356
static const long BAND_CONV = 16777216;
static const long KV_OFF = 66846720;               // 510*HW*B (kv 384ch, later gated 510ch)
static const long XN_OFF = 100270080;              // 192*HW*B
static const long Q_OFF = 112852992;               // 192*HW*B
static const long T_OFF = 125435904;               // 192*HW*B
static const long RQ_OFF = 138018816;
static const long RK_OFF = 138019840;
static const long ATTN_OFF = 138020864;            // 16*48*48

extern "C" void kernel_launch(void* const* d_in, const int* in_sizes, int n_in,
                              void* d_out, int out_size, void* d_ws, size_t ws_size,
                              hipStream_t stream) {
    (void)in_sizes; (void)n_in; (void)out_size; (void)ws_size;
    const float* x        = (const float*)d_in[0];
    const float* ln1_g    = (const float*)d_in[1];
    const float* ln1_b    = (const float*)d_in[2];
    const float* ln2_g    = (const float*)d_in[3];
    const float* ln2_b    = (const float*)d_in[4];
    const float* temp     = (const float*)d_in[5];
    const float* w_qkv    = (const float*)d_in[6];
    const float* w_qkv_dw = (const float*)d_in[7];
    const float* w_ssl5   = (const float*)d_in[8];
    const float* w_ssl7   = (const float*)d_in[9];
    const float* w_ssl9   = (const float*)d_in[10];
    const float* w_attn_o = (const float*)d_in[11];
    const float* w_ffn_in = (const float*)d_in[12];
    const float* w_ffn_dw = (const float*)d_in[13];
    const float* w_ffn_o  = (const float*)d_in[14];
    float* ws  = (float*)d_ws;
    float* out = (float*)d_out;

    // 1) LN1: x -> xn
    ln_kernel<<<256, 256, 0, stream>>>(x, ln1_g, ln1_b, ws + XN_OFF);

    // 2) qkv 1x1 GEMM: (384x192) * (192 x HW) per batch -> bufA
    {
        dim3 g(128, 6, NB);
        gemm_bf16_kernel<<<g, 256, 0, stream>>>(w_qkv, ws + XN_OFF, ws + A_OFF, nullptr,
            384, 192, (int)HW, 0, 0, 192 * HW, 0, 384 * HW, 0, 0, 0, 1);
    }
    // 3) depthwise 3x3 on 2C channels -> kv
    {
        long tot = (long)NB * 384 * HW;
        dw3x3_kernel<<<(unsigned)((tot + 255) / 256), 256, 0, stream>>>(
            ws + A_OFF, w_qkv_dw, ws + KV_OFF, 384, IMG_H, IMG_W, tot);
    }
    // 4) DWT: xn -> 4 bands (reuse bufA region; qkv_out is dead)
    long tot_band = (long)NB * 192 * 66 * 66;
    dwt_kernel<<<(unsigned)((tot_band + 255) / 256), 256, 0, stream>>>(
        ws + XN_OFF, ws + BAND_RAW, tot_band);
    // 5) per-band depthwise 3x3 (w5, w5, w7, w9)
    {
        const float* bw[4] = {w_ssl5, w_ssl5, w_ssl7, w_ssl9};
        for (int bnd = 0; bnd < 4; ++bnd)
            dw3x3_kernel<<<(unsigned)((tot_band + 255) / 256), 256, 0, stream>>>(
                ws + BAND_RAW + (long)bnd * tot_band, bw[bnd],
                ws + BAND_CONV + (long)bnd * tot_band, 192, 66, 66, tot_band);
    }
    // 6) IDWT -> q
    {
        long tot = (long)NB * 192 * HW;
        idwt_kernel<<<(unsigned)((tot + 255) / 256), 256, 0, stream>>>(
            ws + BAND_CONV, ws + Q_OFF, tot, tot_band);
    }
    // 7) reciprocal row norms of q and k
    rownorm_kernel<<<NB * 192, 256, 0, stream>>>(ws + Q_OFF, 192 * HW, 192, ws + RQ_OFF);
    rownorm_kernel<<<NB * 192, 256, 0, stream>>>(ws + KV_OFF, 384 * HW, 192, ws + RK_OFF);
    // 8) Gram + softmax per (b,h)
    gram_softmax_kernel<<<16, 256, 0, stream>>>(ws + Q_OFF, ws + KV_OFF,
        ws + RQ_OFF, ws + RK_OFF, temp, ws + ATTN_OFF);
    // 9) attn @ v per (b,h): (48x48)*(48 x HW) -> bufT
    {
        dim3 g(128, 1, 16);
        gemm_bf16_kernel<<<g, 256, 0, stream>>>(ws + ATTN_OFF, ws + KV_OFF + 192 * HW,
            ws + T_OFF, nullptr, 48, 48, (int)HW,
            4L * 2304, 2304, 384 * HW, 48 * HW, 192 * HW, 48 * HW, 0, 0, 4);
    }
    // 10) attn_out 1x1 GEMM + residual x -> out
    {
        dim3 g(128, 3, NB);
        gemm_bf16_kernel<<<g, 256, 0, stream>>>(w_attn_o, ws + T_OFF, out, x,
            192, 192, (int)HW, 0, 0, 192 * HW, 0, 192 * HW, 0, 192 * HW, 0, 1);
    }
    // 11) LN2: out -> xn2
    ln_kernel<<<256, 256, 0, stream>>>(out, ln2_g, ln2_b, ws + XN_OFF);
    // 12) ffn_in 1x1 GEMM: (1020x192) -> bufA
    {
        dim3 g(128, 16, NB);
        gemm_bf16_kernel<<<g, 256, 0, stream>>>(w_ffn_in, ws + XN_OFF, ws + A_OFF, nullptr,
            1020, 192, (int)HW, 0, 0, 192 * HW, 0, 1020 * HW, 0, 0, 0, 1);
    }
    // 13) fused depthwise + GELU gate -> bufKV (510 ch)
    {
        long tot = (long)NB * 510 * HW;
        gate_kernel<<<(unsigned)((tot + 255) / 256), 256, 0, stream>>>(
            ws + A_OFF, w_ffn_dw, ws + KV_OFF, tot);
    }
    // 14) ffn_out 1x1 GEMM (192x510) + residual out -> out
    {
        dim3 g(128, 3, NB);
        gemm_bf16_kernel<<<g, 256, 0, stream>>>(w_ffn_o, ws + KV_OFF, out, out,
            192, 510, (int)HW, 0, 0, 510 * HW, 0, 192 * HW, 0, 192 * HW, 0, 1);
    }
}
